// FastAttnCondInjection_qk_v_38946763441016
// MI455X (gfx1250) — matrix operand order repaired and checked by reading, not compiled, not run on hardware
//
#include <hip/hip_runtime.h>
#include <hip/hip_bf16.h>

typedef __bf16 bf16_t;
typedef __bf16 v16bf __attribute__((ext_vector_type(16)));
typedef __bf16 v8bf  __attribute__((ext_vector_type(8)));
typedef float  v8f   __attribute__((ext_vector_type(8)));

#define B_   4
#define C_   256
#define H_   128
#define W_   128
#define HW_  (H_ * W_)
#define SCALE_ 0.17677669529663687f   /* 1/sqrt(32) */
#define EPS_   1e-5f

// ---------------------------------------------------------------------------
// 1. GroupNorm: one block per (b,g). Writes channel-first f32 (dwconv input)
//    and channel-last f32 (residual for the ao epilogue).
// ---------------------------------------------------------------------------
__global__ void groupnorm_kernel(const float* __restrict__ x,
                                 const float* __restrict__ gw,
                                 const float* __restrict__ gb,
                                 float* __restrict__ xn_cf,
                                 float* __restrict__ xnT) {
    const int bg = blockIdx.x;            // b*32 + g
    const int g  = bg & 31, bb = bg >> 5;
    const size_t base = ((size_t)bb * C_ + (size_t)g * 8) * HW_;
    const int n = 8 * HW_;                // 131072

    float s = 0.f, ss = 0.f;
    for (int i = threadIdx.x; i < n; i += blockDim.x) {
        float v = x[base + i];
        s += v; ss += v * v;
    }
    __shared__ float rs[256], rq[256];
    rs[threadIdx.x] = s; rq[threadIdx.x] = ss;
    __syncthreads();
    for (int st = 128; st > 0; st >>= 1) {
        if ((int)threadIdx.x < st) {
            rs[threadIdx.x] += rs[threadIdx.x + st];
            rq[threadIdx.x] += rq[threadIdx.x + st];
        }
        __syncthreads();
    }
    const float mean = rs[0] / (float)n;
    const float var  = rq[0] / (float)n - mean * mean;
    const float inv  = rsqrtf(var + EPS_);
    for (int i = threadIdx.x; i < n; i += blockDim.x) {
        int c  = g * 8 + i / HW_;
        int px = i - (i / HW_) * HW_;
        float v = (x[base + i] - mean) * inv * gw[c] + gb[c];
        xn_cf[base + i] = v;
        xnT[((size_t)bb * HW_ + px) * C_ + c] = v;
    }
}

// ---------------------------------------------------------------------------
// 2. Weight packer: f32 [OC][IC][taps] -> bf16 fragment order
//    packed[(((mt*taps+tap)*KS+ks)*32+lane)*16+e], k = 16*(lane/16) + e (contiguous order).
// ---------------------------------------------------------------------------
__global__ void packA_kernel(const float* __restrict__ Wt,
                             bf16_t* __restrict__ packed,
                             int OC, int IC, int taps) {
    size_t idx = (size_t)blockIdx.x * 256 + threadIdx.x;
    size_t total = (size_t)OC * IC * taps;
    if (idx >= total) return;
    int e    = (int)(idx & 15);
    int lane = (int)((idx >> 4) & 31);
    size_t rest = idx >> 9;
    int KS  = IC / 32;
    int ks  = (int)(rest % KS); rest /= KS;
    int tap = (int)(rest % taps); rest /= taps;
    int mt  = (int)rest;
    int hi = lane >> 4, col = lane & 15;
    int kk = hi * 16 + e;
    int oc = mt * 16 + col, ic = ks * 32 + kk;
    packed[idx] = (bf16_t)Wt[((size_t)oc * IC + ic) * taps + tap];
}

// ---------------------------------------------------------------------------
// 3. Depthwise 3x3 (pad=1): f32 channel-first in -> bf16 channel-last out.
// ---------------------------------------------------------------------------
__global__ void dwconv3x3_cl_kernel(const float* __restrict__ in,
                                    const float* __restrict__ dw,
                                    bf16_t* __restrict__ outT) {
    const int idx = blockIdx.x * 256 + threadIdx.x;   // b*HW + n
    const int bz = idx >> 14;
    const int n  = idx & (HW_ - 1);
    const int h = n >> 7, w = n & (W_ - 1);

    int   off[9];
    bool  ok[9];
#pragma unroll
    for (int t = 0; t < 9; ++t) {
        int dy = t / 3 - 1, dx = t % 3 - 1;
        off[t] = dy * W_ + dx;
        ok[t]  = ((unsigned)(h + dy) < H_) && ((unsigned)(w + dx) < W_);
    }
    const size_t ibase = (size_t)bz * C_ * HW_ + n;
    const size_t obase = (size_t)idx * C_;
    for (int c0 = 0; c0 < C_; c0 += 8) {
        v8bf r;
#pragma unroll
        for (int j = 0; j < 8; ++j) {
            int c = c0 + j;
            const float* wp = dw + c * 9;
            const float* ip = in + ibase + (size_t)c * HW_;
            float acc = 0.f;
#pragma unroll
            for (int t = 0; t < 9; ++t)
                if (ok[t]) acc += wp[t] * ip[off[t]];
            r[j] = (bf16_t)acc;
        }
        *(v8bf*)&outT[obase + c0] = r;
    }
}

// ---------------------------------------------------------------------------
// 4. Channel-last implicit GEMM, compile-time specialized on IC / TAPS / SILU.
//    actT: bf16 [B, HW, IC]; packA: fragment-ordered bf16 weights.
//    grid (HW/64, OC/128, B); 8 waves; wave = 16(M) x 64(N).
//    B tile (64 pixels x IC) is staged per tap in LDS in K-contiguous order:
//    cooperative fill = 16B global loads + ds_store_b128; fragment reads are
//    32B-aligned contiguous ds_load_b128 pairs. A comes from packed weights
//    as one contiguous 32B load per lane.
// ---------------------------------------------------------------------------
template <int IC, int TAPS, int SILU>
__global__ __launch_bounds__(256)
void gemm_cl_kernel(const bf16_t* __restrict__ actT,
                    const bf16_t* __restrict__ packA,
                    const float* __restrict__ bias,
                    const float* __restrict__ residT,
                    float* __restrict__ outT_f,
                    bf16_t* __restrict__ outT_b,
                    float* __restrict__ out_cf,
                    int OC) {
    constexpr int KS  = IC / 32;
    constexpr int NCH = (64 * IC) / (256 * 8);   // fill chunks per thread
    __shared__ __align__(32) bf16_t sB[64 * IC]; // 32/64 KB, K contiguous

    const int tid  = threadIdx.x;
    const int lane = tid & 31, wave = tid >> 5;
    const int hi = lane >> 4, col = lane & 15;
    const int n0 = blockIdx.x * 64;
    const int h  = n0 >> 7;
    const int wb = n0 & (W_ - 1);
    const int mt = blockIdx.y * 8 + wave;             // global 16-row M tile
    const int m0 = mt * 16;
    const int bz = blockIdx.z;
    const size_t actB = (size_t)bz * HW_ * IC;
    const bf16_t* apBase = packA + (size_t)mt * TAPS * KS * 512 + lane * 16;

    v8f acc[4] = {};
#pragma unroll
    for (int tap = 0; tap < TAPS; ++tap) {
        const int dy = (TAPS == 9) ? (tap / 3 - 1) : 0;   // constant-folded
        const int dx = (TAPS == 9) ? (tap % 3 - 1) : 0;
        const int hh = h + dy;
        if (TAPS == 9 && (unsigned)hh >= H_) continue;    // block-uniform skip

        // ---- cooperative LDS fill (border rows zeroed) ----
#pragma unroll
        for (int j = 0; j < NCH; ++j) {
            const int f = (j * 256 + tid) * 8;            // flat elem index
            const int nr = f / IC;                        // pixel row 0..63
            const int k  = f & (IC - 1);
            const int ww = wb + nr + dx;
            v8bf val = {};
            if ((TAPS == 1) || ((unsigned)ww < W_))
                val = *(const v8bf*)(actT + actB + (size_t)(hh * W_ + ww) * IC + k);
            *(v8bf*)&sB[nr * IC + k] = val;
        }
        __syncthreads();

        const bf16_t* ap = apBase + (size_t)tap * KS * 512;
        if constexpr (TAPS == 1) {
#pragma unroll
            for (int ks = 0; ks < KS; ++ks) {
                v16bf a = *(const v16bf*)(ap + ks * 512);
#pragma unroll
                for (int t = 0; t < 4; ++t) {
                    v16bf b = *(const v16bf*)&sB[(t * 16 + col) * IC + ks * 32 + hi * 16];
                    acc[t] = __builtin_amdgcn_wmma_f32_16x16x32_bf16(
                        false, a, false, b, (short)0, acc[t], false, false);
                }
            }
        } else {
#pragma unroll 2
            for (int ks = 0; ks < KS; ++ks) {
                v16bf a = *(const v16bf*)(ap + ks * 512);
#pragma unroll
                for (int t = 0; t < 4; ++t) {
                    v16bf b = *(const v16bf*)&sB[(t * 16 + col) * IC + ks * 32 + hi * 16];
                    acc[t] = __builtin_amdgcn_wmma_f32_16x16x32_bf16(
                        false, a, false, b, (short)0, acc[t], false, false);
                }
            }
        }
        __syncthreads();
    }
    // epilogue: per lane, 8 channel-contiguous results at m0 + hi*8 .. +7
#pragma unroll
    for (int t = 0; t < 4; ++t) {
        const int n = n0 + t * 16 + col;
        const size_t rowo = ((size_t)bz * HW_ + n) * OC + m0 + hi * 8;
        float v[8];
#pragma unroll
        for (int r = 0; r < 8; ++r) {
            float f = acc[t][r];
            if (bias) f += bias[m0 + hi * 8 + r];
            if (residT) f += residT[rowo + r];
            if (SILU) f = f / (1.f + __expf(-f));
            v[r] = f;
        }
        if (outT_f) {
#pragma unroll
            for (int r = 0; r < 8; ++r) outT_f[rowo + r] = v[r];
        }
        if (outT_b) {
            v8bf o;
#pragma unroll
            for (int r = 0; r < 8; ++r) o[r] = (bf16_t)v[r];
            *(v8bf*)&outT_b[rowo] = o;
        }
        if (out_cf) {
#pragma unroll
            for (int r = 0; r < 8; ++r)
                out_cf[((size_t)bz * OC + m0 + hi * 8 + r) * HW_ + n] = v[r];
        }
    }
}

// ---------------------------------------------------------------------------
// 5. Softmax over H (k = qkT channels [256,512)). Block per (b,w), thread per
//    channel; fully coalesced 512B rows, per-thread reduction over h.
// ---------------------------------------------------------------------------
__global__ void softmax_h_cl_kernel(const bf16_t* __restrict__ qkT,
                                    bf16_t* __restrict__ ksmT) {
    const int bw = blockIdx.x;            // b*128 + w
    const int bb = bw >> 7, w = bw & 127;
    const int c = threadIdx.x;            // 0..255
    const size_t ib = (size_t)bb * HW_ * 512 + 256 + c;
    const size_t ob = (size_t)bb * HW_ * 256 + c;
    float mx = -1e30f;
    for (int hh = 0; hh < H_; ++hh)
        mx = fmaxf(mx, (float)qkT[ib + (size_t)(hh * W_ + w) * 512]);
    float s = 0.f;
    for (int hh = 0; hh < H_; ++hh)
        s += __expf((float)qkT[ib + (size_t)(hh * W_ + w) * 512] - mx);
    const float inv = 1.f / s;
    for (int hh = 0; hh < H_; ++hh)
        ksmT[ob + (size_t)(hh * W_ + w) * 256] =
            (bf16_t)(__expf((float)qkT[ib + (size_t)(hh * W_ + w) * 512] - mx) * inv);
}

// ---------------------------------------------------------------------------
// 6. Softmax over W (v path), 1/sqrt(d) folded in. Block per (b,h).
// ---------------------------------------------------------------------------
__global__ void softmax_w_cl_kernel(const bf16_t* __restrict__ vT,
                                    bf16_t* __restrict__ vsmT, float scale) {
    const int bh = blockIdx.x;            // b*128 + h
    const int bb = bh >> 7, h = bh & 127;
    const int c = threadIdx.x;
    const size_t base = ((size_t)bb * HW_ + (size_t)h * W_) * 256 + c;
    float mx = -1e30f;
    for (int w = 0; w < W_; ++w)
        mx = fmaxf(mx, (float)vT[base + (size_t)w * 256]);
    float s = 0.f;
    for (int w = 0; w < W_; ++w)
        s += __expf((float)vT[base + (size_t)w * 256] - mx);
    const float inv = scale / s;
    for (int w = 0; w < W_; ++w)
        vsmT[base + (size_t)w * 256] =
            (bf16_t)(__expf((float)vT[base + (size_t)w * 256] - mx) * inv);
}

// ---------------------------------------------------------------------------
// 7. context[bh][d][e] = sum_n q[d,n]*k_sm[e,n].  One block (4 waves) per
//    (b,head); contraction over n (the strided axis here) -- tiny kernel.
// ---------------------------------------------------------------------------
__global__ void context_cl_kernel(const bf16_t* __restrict__ qkT,
                                  const bf16_t* __restrict__ ksmT,
                                  float* __restrict__ ctx) {
    const int bh = blockIdx.x;
    const int bb = bh >> 3, hd = bh & 7;
    const int lane = threadIdx.x & 31, wave = threadIdx.x >> 5;
    const int Mt = wave >> 1, Nt = wave & 1;
    const int hi = lane >> 4, col = lane & 15;
    const bf16_t* qp = qkT + (size_t)bb * HW_ * 512 + hd * 32 + Mt * 16 + col;
    const bf16_t* kp = ksmT + (size_t)bb * HW_ * 256 + hd * 32 + Nt * 16 + col;
    v8f acc = {};
    for (int n0 = 0; n0 < HW_; n0 += 32) {
        v16bf a, b;
#pragma unroll
        for (int e = 0; e < 16; ++e) {
            int kk = hi * 16 + e;
            a[e] = qp[(size_t)(n0 + kk) * 512];
        }
#pragma unroll
        for (int e = 0; e < 16; ++e)
            b[e] = kp[(size_t)(n0 + hi * 16 + e) * 256];
        acc = __builtin_amdgcn_wmma_f32_16x16x32_bf16(
            false, a, false, b, (short)0, acc, false, false);
    }
#pragma unroll
    for (int r = 0; r < 8; ++r)
        ctx[(size_t)bh * 1024 + (Mt * 16 + r + hi * 8) * 32 + Nt * 16 + col] = acc[r];
}

// ---------------------------------------------------------------------------
// 8. out[b,n,hd*32+e] = sum_d ctx[bh][d][e] * v_sm[b,n,hd*32+d].  K=32; the
//    B fragment and the output are channel-contiguous (vector ops).
// ---------------------------------------------------------------------------
__global__ void attn_out_cl_kernel(const float* __restrict__ ctx,
                                   const bf16_t* __restrict__ vsmT,
                                   bf16_t* __restrict__ outT) {
    const int bh = blockIdx.y;
    const int bb = bh >> 3, hd = bh & 7;
    const int lane = threadIdx.x & 31, wave = threadIdx.x >> 5;
    const int Mt = wave & 1, Nt = wave >> 1;
    const int hi = lane >> 4, col = lane & 15;
    const int n = blockIdx.x * 64 + Nt * 16 + col;
    v16bf a;
#pragma unroll
    for (int e = 0; e < 16; ++e) {        // A(M=e,K=d) = ctx[d][e], tiny/L0
        int kk = hi * 16 + e;
        a[e] = (bf16_t)ctx[(size_t)bh * 1024 + kk * 32 + Mt * 16 + col];
    }
    v16bf b = *(const v16bf*)(vsmT + ((size_t)bb * HW_ + n) * 256 + hd * 32 + hi * 16);
    v8f acc = {};
    acc = __builtin_amdgcn_wmma_f32_16x16x32_bf16(
        false, a, false, b, (short)0, acc, false, false);
    v8bf o;
#pragma unroll
    for (int r = 0; r < 8; ++r) o[r] = (bf16_t)acc[r];
    *(v8bf*)&outT[((size_t)bb * HW_ + n) * 256 + hd * 32 + Mt * 16 + hi * 8] = o;
}

// ---------------------------------------------------------------------------
extern "C" void kernel_launch(void* const* d_in, const int* in_sizes, int n_in,
                              void* d_out, int out_size, void* d_ws, size_t ws_size,
                              hipStream_t stream) {
    const float* x       = (const float*)d_in[0];
    const float* cond    = (const float*)d_in[1];
    const float* gn_w    = (const float*)d_in[2];
    const float* gn_b    = (const float*)d_in[3];
    const float* qk_dw   = (const float*)d_in[4];
    const float* qk_pw_w = (const float*)d_in[5];
    const float* qk_pw_b = (const float*)d_in[6];
    const float* v_dw    = (const float*)d_in[7];
    const float* v_pw_w  = (const float*)d_in[8];
    const float* v_pw_b  = (const float*)d_in[9];
    const float* ao_w    = (const float*)d_in[10];
    const float* ao_b    = (const float*)d_in[11];
    const float* ffn_w1  = (const float*)d_in[12];
    const float* ffn_w2  = (const float*)d_in[13];
    const float* ffn_w3  = (const float*)d_in[14];
    const float* ffn_b3  = (const float*)d_in[15];
    float* out = (float*)d_out;

    char* ws = (char*)d_ws;
    const size_t MB = 1024ull * 1024ull;
    float*  xn_cf    = (float*)(ws + 0);          //  64 MB  [B,C,HW] f32
    float*  xnT      = (float*)(ws + 64 * MB);    //  64 MB  [B,HW,C] f32
    float*  aresT_f  = (float*)(ws + 128 * MB);   //  64 MB  [B,HW,C] f32
    bf16_t* qkT      = (bf16_t*)(ws + 192 * MB);  //  64 MB  [B,HW,512]
    bf16_t* t1T      = (bf16_t*)(ws + 256 * MB);  //  32 MB
    bf16_t* t2T      = (bf16_t*)(ws + 288 * MB);  //  32 MB
    bf16_t* vT       = (bf16_t*)(ws + 320 * MB);  //  32 MB
    bf16_t* outAT    = (bf16_t*)(ws + 352 * MB);  //  32 MB
    bf16_t* aresT_b  = (bf16_t*)(ws + 384 * MB);  //  32 MB
    bf16_t* h2T      = (bf16_t*)(ws + 416 * MB);  //  32 MB
    float*  ctx      = (float*)(ws + 448 * MB);   // 128 KB
    bf16_t* paQK     = (bf16_t*)(ws + 449 * MB);  // packed weights (~5.3 MB)
    bf16_t* paV      = paQK + 131072;
    bf16_t* paAO     = paV + 65536;
    bf16_t* paW1     = paAO + 65536;
    bf16_t* paW2     = paW1 + 1179648;
    bf16_t* paW3     = paW2 + 1179648;
    bf16_t* ksmT = t1T;                           // alias: t1 dead after qk GEMM
    bf16_t* vsmT = t2T;                           // alias: t2 dead after v GEMM
    bf16_t* h1T  = qkT;                           // alias: qk dead after context

    // weight packing (tiny)
    packA_kernel<<<512, 256, 0, stream>>>(qk_pw_w, paQK, 512, 256, 1);
    packA_kernel<<<256, 256, 0, stream>>>(v_pw_w, paV, 256, 256, 1);
    packA_kernel<<<256, 256, 0, stream>>>(ao_w, paAO, 256, 256, 1);
    packA_kernel<<<4608, 256, 0, stream>>>(ffn_w1, paW1, 512, 256, 9);
    packA_kernel<<<4608, 256, 0, stream>>>(ffn_w2, paW2, 256, 512, 9);
    packA_kernel<<<256, 256, 0, stream>>>(ffn_w3, paW3, 256, 256, 1);

    groupnorm_kernel<<<128, 256, 0, stream>>>(x, gn_w, gn_b, xn_cf, xnT);
    dwconv3x3_cl_kernel<<<256, 256, 0, stream>>>(cond, qk_dw, t1T);
    dwconv3x3_cl_kernel<<<256, 256, 0, stream>>>(xn_cf, v_dw, t2T);

    // qk = pointwise(t1): OC=512
    gemm_cl_kernel<256, 1, 0><<<dim3(256, 4, 4), 256, 0, stream>>>(
        t1T, paQK, qk_pw_b, nullptr, nullptr, qkT, nullptr, 512);
    // v = pointwise(t2): OC=256
    gemm_cl_kernel<256, 1, 0><<<dim3(256, 2, 4), 256, 0, stream>>>(
        t2T, paV, v_pw_b, nullptr, nullptr, vT, nullptr, 256);

    softmax_h_cl_kernel<<<512, 256, 0, stream>>>(qkT, ksmT);
    softmax_w_cl_kernel<<<512, 256, 0, stream>>>(vT, vsmT, SCALE_);
    context_cl_kernel<<<32, 128, 0, stream>>>(qkT, ksmT, ctx);
    attn_out_cl_kernel<<<dim3(256, 32), 256, 0, stream>>>(ctx, vsmT, outAT);

    // attn_res = ao(out_attn) + bias + xn  (f32 + bf16, channel-last)
    gemm_cl_kernel<256, 1, 0><<<dim3(256, 2, 4), 256, 0, stream>>>(
        outAT, paAO, ao_b, xnT, aresT_f, aresT_b, nullptr, 256);
    // h1 = silu(conv3x3(attn_res, w1)) : 256 -> 512
    gemm_cl_kernel<256, 9, 1><<<dim3(256, 4, 4), 256, 0, stream>>>(
        aresT_b, paW1, nullptr, nullptr, nullptr, h1T, nullptr, 512);
    // h2 = conv3x3(h1, w2) : 512 -> 256
    gemm_cl_kernel<512, 9, 0><<<dim3(256, 2, 4), 256, 0, stream>>>(
        h1T, paW2, nullptr, nullptr, nullptr, h2T, nullptr, 256);
    // out = pointwise(h2, w3) + b3 + attn_res   (channel-first final store)
    gemm_cl_kernel<256, 1, 0><<<dim3(256, 2, 4), 256, 0, stream>>>(
        h2T, paW3, ffn_b3, aresT_f, nullptr, nullptr, out, 256);
}
